// MultiHeadAttention_34651796144182
// MI455X (gfx1250) — matrix operand order repaired and checked by reading, not compiled, not run on hardware
//
#include <hip/hip_runtime.h>
#include <math.h>

// ---------------------------------------------------------------------------
// CDNA5 (gfx1250) bf16 WMMA types & helpers
// ---------------------------------------------------------------------------
typedef __attribute__((ext_vector_type(16))) __bf16 v16bf;
typedef __attribute__((ext_vector_type(8)))  __bf16 v8bf;
typedef __attribute__((ext_vector_type(8)))  float  v8f;

union BF16x16 { v16bf v; v8bf h[2]; };

__device__ __forceinline__ v8f wmma_bf16(v16bf a, v16bf b, v8f c) {
    // (neg_a, A, neg_b, B, c_mod, C, reuse_a, reuse_b)
    return __builtin_amdgcn_wmma_f32_16x16x32_bf16(false, a, false, b,
                                                   (short)0, c, false, false);
}

__device__ __forceinline__ void lds_wave_fence() {
    asm volatile("s_wait_dscnt 0" ::: "memory");
}

// A-fragment (16x32 bf16): lane row = l16; K chunks [8h,8h+8) and [16+8h,+8)
__device__ __forceinline__ void load_afrag(BF16x16& a, const __bf16* rowbase, int lh) {
    a.h[0] = *(const v8bf*)(rowbase + 8 * lh);
    a.h[1] = *(const v8bf*)(rowbase + 16 + 8 * lh);
}
// B-fragment (32x16 bf16): lane col = l16; K chunks [8h,8h+8) and [16+8h,+8)
__device__ __forceinline__ void load_bfrag(BF16x16& b, const __bf16* colbase, int lh) {
    b.h[0] = *(const v8bf*)(colbase + 8 * lh);
    b.h[1] = *(const v8bf*)(colbase + 16 + 8 * lh);
}

// -------- register staging (global -> VGPR -> LDS-as-bf16), 16 elems --------
template <typename T> struct StageReg;
template <> struct StageReg<float>  { float4 f[4]; };
template <> struct StageReg<__bf16> { v8bf v[2]; };

__device__ __forceinline__ void stage_load(StageReg<float>& r, const float* __restrict__ src) {
    const float4* s4 = (const float4*)src;
#pragma unroll
    for (int i = 0; i < 4; ++i) r.f[i] = s4[i];
}
__device__ __forceinline__ void stage_store(const StageReg<float>& r, __bf16* dst) {
#pragma unroll
    for (int hh = 0; hh < 2; ++hh) {
        float4 f0 = r.f[2 * hh + 0];
        float4 f1 = r.f[2 * hh + 1];
        v8bf o;
        o[0] = (__bf16)f0.x; o[1] = (__bf16)f0.y; o[2] = (__bf16)f0.z; o[3] = (__bf16)f0.w;
        o[4] = (__bf16)f1.x; o[5] = (__bf16)f1.y; o[6] = (__bf16)f1.z; o[7] = (__bf16)f1.w;
        *(v8bf*)(dst + 8 * hh) = o;
    }
}
__device__ __forceinline__ void stage_load(StageReg<__bf16>& r, const __bf16* __restrict__ src) {
    const v8bf* s8 = (const v8bf*)src;
    r.v[0] = s8[0];
    r.v[1] = s8[1];
}
__device__ __forceinline__ void stage_store(const StageReg<__bf16>& r, __bf16* dst) {
    *(v8bf*)(dst + 0) = r.v[0];
    *(v8bf*)(dst + 8) = r.v[1];
}

__device__ __forceinline__ void store_out(float* p, float v)  { *p = v; }
__device__ __forceinline__ void store_out(__bf16* p, float v) { *p = (__bf16)v; }

// ---------------------------------------------------------------------------
// GEMM: C[M,N] = A[M,K] * W[N,K]^T + bias[N]
// 256 threads = 8 waves, tile 128x128x32, register-staged double buffering:
// one barrier per K-step; global loads for tile k+1 overlap WMMAs of tile k.
// ---------------------------------------------------------------------------
template <typename TA, typename TC>
__global__ __launch_bounds__(256)
void gemm_bias_wmma(const TA* __restrict__ A, const float* __restrict__ W,
                    const float* __restrict__ bias, TC* __restrict__ C,
                    int M, int N, int K)
{
    constexpr int LDT = 40;                 // padded LDS row stride (bf16 elems)
    __shared__ __align__(16) __bf16 sA[2][128 * LDT];
    __shared__ __align__(16) __bf16 sB[2][128 * LDT];

    const int tid  = threadIdx.x;
    const int wid  = tid >> 5;
    const int lane = tid & 31;
    const int l16  = lane & 15;
    const int lh   = lane >> 4;

    const int bn = blockIdx.x;              // N / 128
    const int bm = blockIdx.y;              // M / 128
    const int wave_m = wid >> 1;            // 0..3
    const int wave_n = wid & 1;             // 0..1

    const int lrow = tid >> 1;              // 0..127 (loader row)
    const int lcol = (tid & 1) * 16;        // loader col: 0 or 16
    const int ldst = lrow * LDT + lcol;

    const TA*    aptr = A + (size_t)(bm * 128 + lrow) * K + lcol;
    const float* wptr = W + (size_t)(bn * 128 + lrow) * K + lcol;

    // prologue: stage tile 0
    {
        StageReg<TA>    ra;
        StageReg<float> rw;
        stage_load(ra, aptr);
        stage_load(rw, wptr);
        stage_store(ra, &sA[0][ldst]);
        stage_store(rw, &sB[0][ldst]);
    }
    __syncthreads();

    v8f acc[2][4];
#pragma unroll
    for (int tm = 0; tm < 2; ++tm)
#pragma unroll
        for (int tn = 0; tn < 4; ++tn)
#pragma unroll
            for (int r = 0; r < 8; ++r) acc[tm][tn][r] = 0.0f;

    int cur = 0;
    for (int k0 = 0; k0 < K; k0 += 32) {
        const bool more = (k0 + 32) < K;
        StageReg<TA>    ra;
        StageReg<float> rw;
        if (more) {
            stage_load(ra, aptr + k0 + 32);          // overlaps WMMAs below
            stage_load(rw, wptr + k0 + 32);
            if (k0 + 64 < K) {                       // warm L2/WGP$ two tiles out
                __builtin_prefetch(aptr + k0 + 64, 0, 0);
                __builtin_prefetch(wptr + k0 + 64, 0, 0);
            }
        }

        const __bf16* cA = sA[cur];
        const __bf16* cB = sB[cur];

        BF16x16 afr[2];
#pragma unroll
        for (int tm = 0; tm < 2; ++tm)
            load_afrag(afr[tm], cA + (wave_m * 32 + tm * 16 + l16) * LDT, lh);
#pragma unroll
        for (int tn = 0; tn < 4; ++tn) {
            BF16x16 bfr;
            load_bfrag(bfr, cB + (wave_n * 64 + tn * 16 + l16) * LDT, lh);
#pragma unroll
            for (int tm = 0; tm < 2; ++tm)
                acc[tm][tn] = wmma_bf16(afr[tm].v, bfr.v, acc[tm][tn]);
        }

        if (more) {
            stage_store(ra, &sA[cur ^ 1][ldst]);
            stage_store(rw, &sB[cur ^ 1][ldst]);
        }
        __syncthreads();
        cur ^= 1;
    }

#pragma unroll
    for (int tm = 0; tm < 2; ++tm)
#pragma unroll
        for (int tn = 0; tn < 4; ++tn) {
            const int col = bn * 128 + wave_n * 64 + tn * 16 + l16;
            const float bv = bias[col];
#pragma unroll
            for (int r = 0; r < 8; ++r) {
                const int row = bm * 128 + wave_m * 32 + tm * 16 + r + 8 * lh;
                store_out(C + (size_t)row * N + col, acc[tm][tn][r] + bv);
            }
        }
}

// ---------------------------------------------------------------------------
// Flash attention (causal) per (b,h). Heads are contiguous [S=2048, DK=64]
// bf16 matrices at offset (b*H+h)*S*DK (raw-reshape semantics for free).
// 128 threads = 4 waves; each wave owns 16 query rows; 64 keys per iteration.
// ---------------------------------------------------------------------------
__global__ __launch_bounds__(128)
void attn_wmma(const __bf16* __restrict__ qb, const __bf16* __restrict__ kb,
               const __bf16* __restrict__ vb, __bf16* __restrict__ ctx)
{
    constexpr int S = 2048, DK = 64, H = 16, D = 1024;
    constexpr int LVT = 72;                 // padded LDS strides
    __shared__ __align__(16) __bf16 sVt[64 * LVT];       // V^T (dk-major)
    __shared__ __align__(16) __bf16 sP[4][16 * LVT];     // per-wave P tile

    const int tid  = threadIdx.x;
    const int wid  = tid >> 5;
    const int lane = tid & 31;
    const int l16  = lane & 15;
    const int lh   = lane >> 4;

    const int mblk = blockIdx.x;            // S/64 query blocks
    const int bh   = blockIdx.y;            // b*H + h
    const int b    = bh >> 4;
    const int h    = bh & (H - 1);

    const size_t headoff = (size_t)bh * S * DK;
    const __bf16* qh = qb + headoff;
    const __bf16* kh = kb + headoff;
    const __bf16* vh = vb + headoff;

    const int m0 = mblk * 64 + wid * 16;    // this wave's first query row (s')

    // Q A-fragments (DK=64 -> two K=32 steps), contiguous global loads
    BF16x16 aq[2];
#pragma unroll
    for (int ks = 0; ks < 2; ++ks)
        load_afrag(aq[ks], qh + (size_t)(m0 + l16) * DK + 32 * ks, lh);

    float mrow[8], lrow[8];
    v8f acc[4];
#pragma unroll
    for (int r = 0; r < 8; ++r) { mrow[r] = -3.0e38f; lrow[r] = 0.0f; }
#pragma unroll
    for (int tn = 0; tn < 4; ++tn)
#pragma unroll
        for (int r = 0; r < 8; ++r) acc[tn][r] = 0.0f;

    for (int j = 0; j <= mblk; ++j) {
        const int n0 = j * 64;

        // cooperative V^T staging (transpose scatter): 128 threads x 32 elems
        {
            const int kr  = tid >> 1;            // key row 0..63
            const int dk0 = (tid & 1) * 32;
            const __bf16* vsrc = vh + (size_t)(n0 + kr) * DK + dk0;
#pragma unroll
            for (int i = 0; i < 32; ++i)
                sVt[(dk0 + i) * LVT + kr] = vsrc[i];
            if (j < mblk) {                      // warm caches for next block
                __builtin_prefetch(vh + (size_t)(n0 + 64 + kr) * DK + dk0, 0, 0);
                __builtin_prefetch(kh + (size_t)(n0 + 64 + kr) * DK + dk0, 0, 0);
            }
        }
        __syncthreads();

        // scores S = Q K^T : key rows contiguous -> direct global B-frags
        v8f sc[4];
#pragma unroll
        for (int tn = 0; tn < 4; ++tn) {
#pragma unroll
            for (int r = 0; r < 8; ++r) sc[tn][r] = 0.0f;
#pragma unroll
            for (int ks = 0; ks < 2; ++ks) {
                BF16x16 bk;
                load_bfrag(bk, kh + (size_t)(n0 + tn * 16 + l16) * DK + 32 * ks, lh);
                sc[tn] = wmma_bf16(aq[ks].v, bk.v, sc[tn]);
            }
        }

        // scale + causal mask (C layout: lane col = n0+tn*16+l16, row = m0+r+8*lh)
#pragma unroll
        for (int tn = 0; tn < 4; ++tn) {
            const int col = n0 + tn * 16 + l16;
#pragma unroll
            for (int r = 0; r < 8; ++r) {
                float s = sc[tn][r] * 0.125f;   // 1/sqrt(64)
                const int row = m0 + r + 8 * lh;
                if (col > row) s = -1.0e9f;
                sc[tn][r] = s;
            }
        }

        // online softmax: row max via 16-lane butterfly
        float bmax[8];
#pragma unroll
        for (int r = 0; r < 8; ++r) {
            float v = sc[0][r];
#pragma unroll
            for (int tn = 1; tn < 4; ++tn) v = fmaxf(v, sc[tn][r]);
            bmax[r] = v;
        }
#pragma unroll
        for (int msk = 1; msk <= 8; msk <<= 1)
#pragma unroll
            for (int r = 0; r < 8; ++r)
                bmax[r] = fmaxf(bmax[r], __shfl_xor(bmax[r], msk, 32));

        float corr[8];
#pragma unroll
        for (int r = 0; r < 8; ++r) {
            const float mn = fmaxf(mrow[r], bmax[r]);
            corr[r] = __expf(mrow[r] - mn);
            mrow[r] = mn;
        }

        // P = exp(S - m): accumulate row sums, stage P tile (row-major 16x64)
        float bsum[8];
#pragma unroll
        for (int r = 0; r < 8; ++r) bsum[r] = 0.0f;
#pragma unroll
        for (int tn = 0; tn < 4; ++tn)
#pragma unroll
            for (int r = 0; r < 8; ++r) {
                const float p = __expf(sc[tn][r] - mrow[r]);
                bsum[r] += p;
                sP[wid][(r + 8 * lh) * LVT + tn * 16 + l16] = (__bf16)p;
            }
#pragma unroll
        for (int msk = 1; msk <= 8; msk <<= 1)
#pragma unroll
            for (int r = 0; r < 8; ++r)
                bsum[r] += __shfl_xor(bsum[r], msk, 32);
#pragma unroll
        for (int r = 0; r < 8; ++r) lrow[r] = lrow[r] * corr[r] + bsum[r];
#pragma unroll
        for (int tn = 0; tn < 4; ++tn)
#pragma unroll
            for (int r = 0; r < 8; ++r) acc[tn][r] *= corr[r];

        lds_wave_fence();   // wave-local P store -> A-frag load ordering

        // PV: A = P (from LDS), B = V^T rows (dk-major, contiguous K span)
        BF16x16 pa[2];
#pragma unroll
        for (int ks = 0; ks < 2; ++ks)
            load_afrag(pa[ks], &sP[wid][l16 * LVT + 32 * ks], lh);
#pragma unroll
        for (int tn = 0; tn < 4; ++tn) {
#pragma unroll
            for (int ks = 0; ks < 2; ++ks) {
                BF16x16 vf;
                load_bfrag(vf, &sVt[(tn * 16 + l16) * LVT + 32 * ks], lh);
                acc[tn] = wmma_bf16(pa[ks].v, vf.v, acc[tn]);
            }
        }
        __syncthreads();    // protect sVt before next iteration's staging
    }

    // normalize & scatter to context in [B, S, D] layout (head-interleaved)
#pragma unroll
    for (int tn = 0; tn < 4; ++tn) {
        const int col = h * DK + tn * 16 + l16;
#pragma unroll
        for (int r = 0; r < 8; ++r) {
            const int row = m0 + r + 8 * lh;   // s'
            const float o = acc[tn][r] / lrow[r];
            ctx[(size_t)b * S * D + (size_t)row * D + col] = (__bf16)o;
        }
    }
}

// ---------------------------------------------------------------------------
// Host orchestration
// ---------------------------------------------------------------------------
extern "C" void kernel_launch(void* const* d_in, const int* in_sizes, int n_in,
                              void* d_out, int out_size, void* d_ws, size_t ws_size,
                              hipStream_t stream)
{
    (void)in_sizes; (void)n_in; (void)out_size; (void)ws_size;

    const float* query = (const float*)d_in[0];
    const float* key_  = (const float*)d_in[1];
    const float* value = (const float*)d_in[2];
    // d_in[3] = causal mask (int) -- computed analytically in-kernel
    const float* Wq = (const float*)d_in[4];
    const float* bq = (const float*)d_in[5];
    const float* Wk = (const float*)d_in[6];
    const float* bk = (const float*)d_in[7];
    const float* Wv = (const float*)d_in[8];
    const float* bv = (const float*)d_in[9];
    const float* Wo = (const float*)d_in[10];
    const float* bo = (const float*)d_in[11];
    float* out = (float*)d_out;

    constexpr int B = 2, S = 2048, D = 1024, H = 16;
    constexpr int M = B * S;                 // 4096
    const size_t elems = (size_t)M * D;

    __bf16* qb = (__bf16*)d_ws;              // 8 MB each
    __bf16* kb = qb + elems;
    __bf16* vb = kb + elems;
    __bf16* cb = vb + elems;

    dim3 gG(D / 128, M / 128);               // (8, 32)
    gemm_bias_wmma<float, __bf16><<<gG, 256, 0, stream>>>(query, Wq, bq, qb, M, D, D);
    gemm_bias_wmma<float, __bf16><<<gG, 256, 0, stream>>>(key_,  Wk, bk, kb, M, D, D);
    gemm_bias_wmma<float, __bf16><<<gG, 256, 0, stream>>>(value, Wv, bv, vb, M, D, D);

    dim3 gA(S / 64, B * H);                  // (32, 32)
    attn_wmma<<<gA, 128, 0, stream>>>(qb, kb, vb, cb);

    gemm_bias_wmma<__bf16, float><<<gG, 256, 0, stream>>>(cb, Wo, bo, out, M, D, D);
}
